// BioSSMMixer_60352880443644
// MI455X (gfx1250) — matrix operand order repaired and checked by reading, not compiled, not run on hardware
//
#include <hip/hip_runtime.h>
#include <hip/hip_bf16.h>
#include <math.h>

#define D_   2688
#define B_   2
#define T_   2048
#define TOK  (B_ * T_)        // 4096
#define K_GR 16               // spiking threshold groups
#define BK   64               // K-chunk per LDS stage
#define LDSP 72               // padded LDS row stride in halves (144B, 16B-aligned)

typedef _Float16 v16h __attribute__((ext_vector_type(16)));
typedef _Float16 v8h  __attribute__((ext_vector_type(8)));
typedef float    v8f  __attribute__((ext_vector_type(8)));

union Frag { v16h v; v8h h[2]; };

// ---------------------------------------------------------------------------
// W (D x D, f32, row-major) -> Wt (D x D, f16, N-major: Wt[n*D + k] = W[k][n])
// ---------------------------------------------------------------------------
__global__ __launch_bounds__(256) void transpose_f16(const float* __restrict__ W,
                                                     _Float16* __restrict__ Wt) {
    __shared__ float tile[32][33];
    const int bx = blockIdx.x, by = blockIdx.y;
    const int tx = threadIdx.x, ty = threadIdx.y;     // (32, 8)
#pragma unroll
    for (int j = 0; j < 32; j += 8)
        tile[ty + j][tx] = W[(size_t)(by * 32 + ty + j) * D_ + bx * 32 + tx];
    __syncthreads();
#pragma unroll
    for (int j = 0; j < 32; j += 8)
        Wt[(size_t)(bx * 32 + ty + j) * D_ + by * 32 + tx] = (_Float16)tile[tx][ty + j];
}

// ---------------------------------------------------------------------------
// LayerNorm per token -> x (f16) ; fused tiny projections Bm = x@W_B, Cm = x@W_C
// ---------------------------------------------------------------------------
__device__ __forceinline__ float blockReduce(float v, float* red) {
    const int t = threadIdx.x;
    red[t] = v; __syncthreads();
    for (int o = 128; o > 0; o >>= 1) {
        if (t < o) red[t] += red[t + o];
        __syncthreads();
    }
    const float r = red[0];
    __syncthreads();
    return r;
}

__global__ __launch_bounds__(256) void ln_proj(const float* __restrict__ h,
                                               const float* __restrict__ gamma,
                                               const float* __restrict__ beta,
                                               const float* __restrict__ WB,
                                               const float* __restrict__ WC,
                                               _Float16* __restrict__ xh,
                                               float* __restrict__ Bm,
                                               float* __restrict__ Cm) {
    __shared__ float red[256];
    const int tok = blockIdx.x;
    const float* hr = h + (size_t)tok * D_;

    float s = 0.f, ss = 0.f;
    for (int i = threadIdx.x; i < D_; i += 256) { float v = hr[i]; s += v; ss += v * v; }
    const float mean = blockReduce(s, red) * (1.f / D_);
    const float var  = blockReduce(ss, red) * (1.f / D_) - mean * mean;
    const float rstd = rsqrtf(var + 1e-5f);

    float pb[4] = {0, 0, 0, 0}, pc[4] = {0, 0, 0, 0};
    for (int i = threadIdx.x; i < D_; i += 256) {
        const float x = (hr[i] - mean) * rstd * gamma[i] + beta[i];
        xh[(size_t)tok * D_ + i] = (_Float16)x;
#pragma unroll
        for (int n = 0; n < 4; n++) { pb[n] += x * WB[i * 4 + n]; pc[n] += x * WC[i * 4 + n]; }
    }
#pragma unroll
    for (int n = 0; n < 4; n++) {
        const float rb = blockReduce(pb[n], red);
        if (threadIdx.x == 0) Bm[tok * 4 + n] = rb;
        const float rc = blockReduce(pc[n], red);
        if (threadIdx.x == 0) Cm[tok * 4 + n] = rc;
    }
}

// ---------------------------------------------------------------------------
// Y(M x D_) = X(M x D_, f16) @ W(D_ x D_)   via  Wt (N-major f16)
// mode: 0 = identity, 1 = sigmoid, 2 = softplus(y + bias[col])
// Block tile 128x128, BK=64, double-buffered LDS (1 barrier / 64-K chunk),
// 8 waves (2x4), wave tile 64x32 = 4x2 WMMA accumulators, b128 global loads.
// ---------------------------------------------------------------------------
__global__ __launch_bounds__(256) void gemm_wmma(const _Float16* __restrict__ X,
                                                 const _Float16* __restrict__ Wt,
                                                 float* __restrict__ Y,
                                                 const float* __restrict__ bias,
                                                 int mode) {
    __shared__ __align__(16) _Float16 As[2][128][LDSP];
    __shared__ __align__(16) _Float16 Bs[2][128][LDSP];

    const int tid   = threadIdx.x;
    const int lane  = tid & 31;
    const int wave  = tid >> 5;
    const int wm    = wave & 1;        // 2 wave-rows
    const int wn    = wave >> 1;       // 4 wave-cols
    const int m0    = blockIdx.y * 128;
    const int n0    = blockIdx.x * 128;
    const int r16   = lane & 15;
    const int khalf = lane >> 4;

    // global staging: 16B per thread per pass, 4 passes per 128x64 tile
    const int grow = tid >> 3;          // 8 threads per row
    const int gcol = (tid & 7) * 8;     // 8 halves per thread

    v8f acc[4][2];
#pragma unroll
    for (int i = 0; i < 4; i++)
#pragma unroll
        for (int j = 0; j < 2; j++)
#pragma unroll
            for (int r = 0; r < 8; r++) acc[i][j][r] = 0.f;

    v8h regA[4], regB[4];
    constexpr int NC = D_ / BK;        // 42 chunks

    // preload chunk 0 -> buffer 0
#pragma unroll
    for (int c = 0; c < 4; c++) {
        const int row = c * 32 + grow;
        regA[c] = *(const v8h*)&X [(size_t)(m0 + row) * D_ + gcol];
        regB[c] = *(const v8h*)&Wt[(size_t)(n0 + row) * D_ + gcol];
    }
#pragma unroll
    for (int c = 0; c < 4; c++) {
        const int row = c * 32 + grow;
        *(v8h*)&As[0][row][gcol] = regA[c];
        *(v8h*)&Bs[0][row][gcol] = regB[c];
    }
    __syncthreads();

    for (int kc = 0; kc < NC; kc++) {
        const int cur = kc & 1;
        const int nxt = cur ^ 1;
        const bool more = (kc + 1) < NC;

        // issue global loads for next chunk before computing current one
        if (more) {
            const int kofs = (kc + 1) * BK;
#pragma unroll
            for (int c = 0; c < 4; c++) {
                const int row = c * 32 + grow;
                regA[c] = *(const v8h*)&X [(size_t)(m0 + row) * D_ + kofs + gcol];
                regB[c] = *(const v8h*)&Wt[(size_t)(n0 + row) * D_ + kofs + gcol];
            }
        }

        // compute on current buffer: 2 x 32-K sub-chunks, 8 WMMA each
#pragma unroll
        for (int ks = 0; ks < BK; ks += 32) {
            const int kbA = ks + khalf * 16;  // A frag: contiguous 16
            const int kbB = ks + khalf * 16;  // B frag: contiguous 16

            Frag b[2];
#pragma unroll
            for (int j = 0; j < 2; j++) {
                const int n = wn * 32 + j * 16 + r16;
                b[j].h[0] = *(const v8h*)&Bs[cur][n][kbB];
                b[j].h[1] = *(const v8h*)&Bs[cur][n][kbB + 8];
            }
#pragma unroll
            for (int i = 0; i < 4; i++) {
                const int m = wm * 64 + i * 16 + r16;
                Frag a;
                a.h[0] = *(const v8h*)&As[cur][m][kbA];
                a.h[1] = *(const v8h*)&As[cur][m][kbA + 8];
#pragma unroll
                for (int j = 0; j < 2; j++)
                    acc[i][j] = __builtin_amdgcn_wmma_f32_16x16x32_f16(
                        false, a.v, false, b[j].v, (short)0, acc[i][j], false, false);
            }
        }

        // stage next chunk into the other buffer
        if (more) {
#pragma unroll
            for (int c = 0; c < 4; c++) {
                const int row = c * 32 + grow;
                *(v8h*)&As[nxt][row][gcol] = regA[c];
                *(v8h*)&Bs[nxt][row][gcol] = regB[c];
            }
        }
        __syncthreads();
    }

    // Epilogue. C/D layout: lane<16 -> rows M=r, lane>=16 -> rows M=r+8; col = lane&15.
#pragma unroll
    for (int i = 0; i < 4; i++) {
        const int rowb = m0 + wm * 64 + i * 16 + khalf * 8;
#pragma unroll
        for (int j = 0; j < 2; j++) {
            const int col = n0 + wn * 32 + j * 16 + r16;
#pragma unroll
            for (int r = 0; r < 8; r++) {
                float y = acc[i][j][r];
                if (mode == 1) {
                    y = 1.f / (1.f + expf(-y));
                } else if (mode == 2) {
                    const float t = y + bias[col];
                    y = (t > 20.f) ? t : log1pf(expf(t));
                }
                Y[(size_t)(rowb + r) * D_ + col] = y;
            }
        }
    }
}

// ---------------------------------------------------------------------------
// Sequential scan over T: one thread per (b, d) channel. Emits g as f16.
// ---------------------------------------------------------------------------
__global__ __launch_bounds__(256) void scan_kernel(const float* __restrict__ u,
                                                   const float* __restrict__ z,
                                                   const float* __restrict__ dt,
                                                   const float* __restrict__ Bm,
                                                   const float* __restrict__ Cm,
                                                   const float* __restrict__ A_log,
                                                   const float* __restrict__ v_th_raw,
                                                   _Float16* __restrict__ gh) {
    const int id = blockIdx.x * 256 + threadIdx.x;
    if (id >= B_ * D_) return;
    const int b = id / D_;
    const int d = id % D_;

    float A[4];
#pragma unroll
    for (int n = 0; n < 4; n++) A[n] = -expf(A_log[d * 4 + n]);

    const float vraw = v_th_raw[d / (D_ / K_GR)];
    const float vth  = 0.1f + ((vraw > 20.f) ? vraw : log1pf(expf(vraw)));

    float s[4] = {0, 0, 0, 0};
    float v = 0.f;
    for (int t = 0; t < T_; t++) {
        const size_t tok = (size_t)b * T_ + t;
        const float dtv = dt[tok * D_ + d];
        const float uv  = u [tok * D_ + d];
        const float zv  = z [tok * D_ + d];
        float y = 0.f;
#pragma unroll
        for (int n = 0; n < 4; n++) {
            const float dec = expf(dtv * A[n]);
            s[n] = dec * s[n] + dtv * Bm[tok * 4 + n] * uv;
            y += s[n] * Cm[tok * 4 + n];
        }
        v = 0.9f * v + y;
        const float spike = 1.f / (1.f + expf(-4.f * (v - vth)));
        v -= spike * vth;
        gh[tok * D_ + d] = (_Float16)(spike * y * zv);
    }
}

// ---------------------------------------------------------------------------
extern "C" void kernel_launch(void* const* d_in, const int* in_sizes, int n_in,
                              void* d_out, int out_size, void* d_ws, size_t ws_size,
                              hipStream_t stream) {
    const float* h        = (const float*)d_in[0];
    const float* gamma    = (const float*)d_in[1];
    const float* beta     = (const float*)d_in[2];
    const float* W_in     = (const float*)d_in[3];
    const float* W_z      = (const float*)d_in[4];
    const float* W_dt     = (const float*)d_in[5];
    const float* b_dt     = (const float*)d_in[6];
    const float* W_B      = (const float*)d_in[7];
    const float* W_C      = (const float*)d_in[8];
    const float* A_log    = (const float*)d_in[9];
    const float* W_out    = (const float*)d_in[10];
    const float* v_th_raw = (const float*)d_in[11];
    float* out = (float*)d_out;

    char*  ws  = (char*)d_ws;
    size_t off = 0;
    auto take = [&](size_t bytes) -> void* {
        void* p = ws + off;
        off = (off + bytes + 255) & ~(size_t)255;
        return p;
    };

    _Float16* xh     = (_Float16*)take((size_t)TOK * D_ * sizeof(_Float16));
    _Float16* gh     = (_Float16*)take((size_t)TOK * D_ * sizeof(_Float16));
    _Float16* wt_in  = (_Float16*)take((size_t)D_ * D_ * sizeof(_Float16));
    _Float16* wt_z   = (_Float16*)take((size_t)D_ * D_ * sizeof(_Float16));
    _Float16* wt_dt  = (_Float16*)take((size_t)D_ * D_ * sizeof(_Float16));
    _Float16* wt_out = (_Float16*)take((size_t)D_ * D_ * sizeof(_Float16));
    float*    u_buf  = (float*)take((size_t)TOK * D_ * sizeof(float));
    float*    z_buf  = (float*)take((size_t)TOK * D_ * sizeof(float));
    float*    dt_buf = (float*)take((size_t)TOK * D_ * sizeof(float));
    float*    Bm     = (float*)take((size_t)TOK * 4 * sizeof(float));
    float*    Cm     = (float*)take((size_t)TOK * 4 * sizeof(float));

    const dim3 tb(32, 8);
    const dim3 tg(D_ / 32, D_ / 32);                 // 84 x 84
    transpose_f16<<<tg, tb, 0, stream>>>(W_in,  wt_in);
    transpose_f16<<<tg, tb, 0, stream>>>(W_z,   wt_z);
    transpose_f16<<<tg, tb, 0, stream>>>(W_dt,  wt_dt);
    transpose_f16<<<tg, tb, 0, stream>>>(W_out, wt_out);

    ln_proj<<<TOK, 256, 0, stream>>>(h, gamma, beta, W_B, W_C, xh, Bm, Cm);

    const dim3 gg(D_ / 128, TOK / 128);              // 21 x 32
    gemm_wmma<<<gg, 256, 0, stream>>>(xh, wt_in, u_buf,  nullptr, 0);
    gemm_wmma<<<gg, 256, 0, stream>>>(xh, wt_z,  z_buf,  nullptr, 1);
    gemm_wmma<<<gg, 256, 0, stream>>>(xh, wt_dt, dt_buf, b_dt,    2);

    scan_kernel<<<(B_ * D_ + 255) / 256, 256, 0, stream>>>(
        u_buf, z_buf, dt_buf, Bm, Cm, A_log, v_th_raw, gh);

    gemm_wmma<<<gg, 256, 0, stream>>>(gh, wt_out, out, nullptr, 0);
}
